// GPT2_47923245089124
// MI455X (gfx1250) — matrix operand order repaired and checked by reading, not compiled, not run on hardware
//
#include <hip/hip_runtime.h>
#include <hip/hip_bf16.h>
#include <math.h>

#define N_EMBD   768
#define N_HEAD   12
#define HEAD_DIM 64
#define N_LAYERS 12
#define VOCAB    50257
#define D_FF     3072
#define BATCH    2
#define SEQ      1024
#define ROWS     (BATCH * SEQ)   // 2048 token rows

typedef __bf16 bf16;
typedef __attribute__((ext_vector_type(16))) __bf16 v16bf;
typedef __attribute__((ext_vector_type(8)))  float  v8f;

// ---------- scalar helpers ----------
static __device__ inline bf16 f2bf(float f) {
  unsigned u = __builtin_bit_cast(unsigned, f);
  unsigned r = u + 0x7FFFu + ((u >> 16) & 1u);   // round-to-nearest-even
  unsigned short h = (unsigned short)(r >> 16);
  return __builtin_bit_cast(bf16, h);
}
static __device__ inline float bf2f(bf16 b) {
  unsigned short h = __builtin_bit_cast(unsigned short, b);
  unsigned u = ((unsigned)h) << 16;
  return __builtin_bit_cast(float, u);
}
static __device__ inline float geluf(float x) {
  return 0.5f * x * (1.0f + erff(x * 0.70710678118654752f));
}

// ---------- WMMA fragment types (16-bit operands; contiguous k order, lane half h holds k 16h..16h+15) ----------
union FragU { v16bf v; uint4 q[2]; };

// ---------- GEMM: one wave32 per block computes a 32(M) x 64(N) C tile ----------
// C = A @ B^T (+bias) (+=Cf) (GELU) -> fp32 Cf and/or bf16 Cb. Batched via blockIdx.z.
// M must be a multiple of 32; N edge handled (vocab).
// causal: 0 = none; 1 = skip tiles fully above diagonal (QK^T); 2 = clamp K to m0+32 (PV).
__global__ __launch_bounds__(32)
void gemm_nt(const bf16* __restrict__ A, long sA, int lda,
             const bf16* __restrict__ B, long sB, int ldb,
             const float* __restrict__ bias,
             float* Cf, long sCf, bf16* Cb, long sCb,
             int ldc, int M, int N, int K, int addC, int act, int causal) {
  int lane = threadIdx.x;
  int n0 = blockIdx.x << 6;           // 64-wide N tile
  int m0 = blockIdx.y << 5;           // 32-tall M tile
  if (causal == 1 && n0 >= m0 + 32) return;       // fully-masked score tile
  if (causal == 2) { int kl = m0 + 32; if (kl < K) K = kl; }  // probs are 0 past row
  long bz = blockIdx.z;
  A += bz * sA;  B += bz * sB;
  if (Cf) Cf += bz * sCf;
  if (Cb) Cb += bz * sCb;

  // Hoisted per-lane row bases (A frag: lanes 0-15 K 0-15, lanes 16-31 K 16-31;
  // B frag: lanes 0-15 K 0-15, lanes 16-31 K 16-31).
  int lm = lane & 15;
  int hi = lane >> 4;
  const bf16* pa0 = A + (size_t)(m0 + lm) * lda + (hi << 4);
  const bf16* pa1 = pa0 + (size_t)16 * lda;
  int nb0 = n0 + lm;                    // clamp rows for the vocab edge (never stored)
  const bf16* pb0 = B + (size_t)((nb0      < N) ? nb0      : N - 1) * ldb + (hi << 4);
  const bf16* pb1 = B + (size_t)((nb0 + 16 < N) ? nb0 + 16 : N - 1) * ldb + (hi << 4);
  const bf16* pb2 = B + (size_t)((nb0 + 32 < N) ? nb0 + 32 : N - 1) * ldb + (hi << 4);
  const bf16* pb3 = B + (size_t)((nb0 + 48 < N) ? nb0 + 48 : N - 1) * ldb + (hi << 4);
  // prefetch bases (one line per lane covers the tile rows)
  const bf16* pfa  = A + (size_t)(m0 + lane) * lda;
  const bf16* pfb0 = B + (size_t)((n0 + lane      < N) ? n0 + lane      : N - 1) * ldb;
  const bf16* pfb1 = B + (size_t)((n0 + 32 + lane < N) ? n0 + 32 + lane : N - 1) * ldb;

  v8f acc[2][4] = {};
  for (int k0 = 0; k0 < K; k0 += 32) {
    // speculative prefetch two K-steps ahead (dropped silently if past the extent)
    __builtin_prefetch(pfa  + k0 + 64, 0, 1);
    __builtin_prefetch(pfb0 + k0 + 64, 0, 1);
    __builtin_prefetch(pfb1 + k0 + 64, 0, 1);

    FragU a0, a1, b0, b1, b2, b3;
    a0.q[0] = *(const uint4*)(pa0 + k0);      a0.q[1] = *(const uint4*)(pa0 + k0 + 8);
    a1.q[0] = *(const uint4*)(pa1 + k0);      a1.q[1] = *(const uint4*)(pa1 + k0 + 8);
    b0.q[0] = *(const uint4*)(pb0 + k0);      b0.q[1] = *(const uint4*)(pb0 + k0 + 8);
    b1.q[0] = *(const uint4*)(pb1 + k0);      b1.q[1] = *(const uint4*)(pb1 + k0 + 8);
    b2.q[0] = *(const uint4*)(pb2 + k0);      b2.q[1] = *(const uint4*)(pb2 + k0 + 8);
    b3.q[0] = *(const uint4*)(pb3 + k0);      b3.q[1] = *(const uint4*)(pb3 + k0 + 8);

    acc[0][0] = __builtin_amdgcn_wmma_f32_16x16x32_bf16(false, a0.v, false, b0.v, (short)0, acc[0][0], false, false);
    acc[1][0] = __builtin_amdgcn_wmma_f32_16x16x32_bf16(false, a1.v, false, b0.v, (short)0, acc[1][0], false, false);
    acc[0][1] = __builtin_amdgcn_wmma_f32_16x16x32_bf16(false, a0.v, false, b1.v, (short)0, acc[0][1], false, false);
    acc[1][1] = __builtin_amdgcn_wmma_f32_16x16x32_bf16(false, a1.v, false, b1.v, (short)0, acc[1][1], false, false);
    acc[0][2] = __builtin_amdgcn_wmma_f32_16x16x32_bf16(false, a0.v, false, b2.v, (short)0, acc[0][2], false, false);
    acc[1][2] = __builtin_amdgcn_wmma_f32_16x16x32_bf16(false, a1.v, false, b2.v, (short)0, acc[1][2], false, false);
    acc[0][3] = __builtin_amdgcn_wmma_f32_16x16x32_bf16(false, a0.v, false, b3.v, (short)0, acc[0][3], false, false);
    acc[1][3] = __builtin_amdgcn_wmma_f32_16x16x32_bf16(false, a1.v, false, b3.v, (short)0, acc[1][3], false, false);
  }

  int rbase = hi << 3;                 // 0 or 8
#pragma unroll
  for (int j = 0; j < 4; ++j) {
    int col = n0 + (j << 4) + lm;
    if (col >= N) continue;
    float bi = bias ? bias[col] : 0.0f;
#pragma unroll
    for (int i = 0; i < 2; ++i) {
#pragma unroll
      for (int r = 0; r < 8; ++r) {
        int row = m0 + (i << 4) + rbase + r;
        long idx = (long)row * ldc + col;
        float v = acc[i][j][r] + bi;
        if (addC) v += Cf[idx];
        if (act)  v = geluf(v);
        if (Cf) Cf[idx] = v;
        if (Cb) Cb[idx] = f2bf(v);
      }
    }
  }
}

// ---------- weight prep ----------
__global__ __launch_bounds__(256)
void cvt_f32_bf16(const float* __restrict__ in, bf16* __restrict__ out, long n) {
  long i = (long)blockIdx.x * 256 + threadIdx.x;
  if (i < n) out[i] = f2bf(in[i]);
}

// Transpose-convert one [K,N] fp32 matrix (per layer, blockIdx.z) to [N,K] bf16.
// K, N multiples of 32 for all uses.
__global__ __launch_bounds__(256)
void transpose_cvt_k(const float* __restrict__ in, bf16* __restrict__ outT,
                     int K, int N) {
  __shared__ float tile[32][33];
  long layerOff = (long)blockIdx.z * K * N;
  const float* src = in + layerOff;
  bf16* dst = outT + layerOff;
  int n0 = blockIdx.x << 5;
  int k0 = blockIdx.y << 5;
  int tx = threadIdx.x & 31;
  int ty = threadIdx.x >> 5;          // 8 rows of threads
#pragma unroll
  for (int r = 0; r < 32; r += 8)
    tile[ty + r][tx] = src[(long)(k0 + ty + r) * N + (n0 + tx)];
  __syncthreads();
#pragma unroll
  for (int r = 0; r < 32; r += 8)
    dst[(long)(n0 + ty + r) * K + (k0 + tx)] = f2bf(tile[tx][ty + r]);
}

// ---------- elementwise / normalization kernels ----------
__global__ __launch_bounds__(256)
void copy_f32(const float* __restrict__ in, float* __restrict__ out, long n) {
  long i = (long)blockIdx.x * 256 + threadIdx.x;
  if (i < n) out[i] = in[i];
}

__global__ __launch_bounds__(256)
void layernorm_k(const float* __restrict__ X, const float* __restrict__ g,
                 const float* __restrict__ b, bf16* __restrict__ out) {
  __shared__ float red[256];
  int row = blockIdx.x;
  int t = threadIdx.x;
  const float* xr = X + (long)row * N_EMBD;
  float s = 0.f, s2 = 0.f;
  for (int i = t; i < N_EMBD; i += 256) { float v = xr[i]; s += v; s2 += v * v; }
  red[t] = s; __syncthreads();
  for (int o = 128; o > 0; o >>= 1) { if (t < o) red[t] += red[t + o]; __syncthreads(); }
  float mean = red[0] * (1.0f / N_EMBD);
  __syncthreads();
  red[t] = s2; __syncthreads();
  for (int o = 128; o > 0; o >>= 1) { if (t < o) red[t] += red[t + o]; __syncthreads(); }
  float var  = red[0] * (1.0f / N_EMBD) - mean * mean;
  float rstd = rsqrtf(var + 1e-5f);
  bf16* orow = out + (long)row * N_EMBD;
  for (int i = t; i < N_EMBD; i += 256)
    orow[i] = f2bf((xr[i] - mean) * rstd * g[i] + b[i]);
}

// qkv[B*S, 3D] bf16 -> q,k [b,h,s,hd] (q scaled by 1/8) and vT [b,h,hd,s]
__global__ __launch_bounds__(256)
void split_qkv_k(const bf16* __restrict__ qkv, bf16* __restrict__ q,
                 bf16* __restrict__ k, bf16* __restrict__ vT) {
  long i = (long)blockIdx.x * 256 + threadIdx.x;
  if (i >= (long)ROWS * N_EMBD) return;
  int  d  = (int)(i % N_EMBD);
  long bs = i / N_EMBD;
  int  s  = (int)(bs % SEQ);
  int  b  = (int)(bs / SEQ);
  int  h  = d / HEAD_DIM, hd = d % HEAD_DIM;
  long bh   = (long)(b * N_HEAD + h);
  long dst  = (bh * SEQ + s) * HEAD_DIM + hd;      // [b,h,s,hd]
  long dstT = (bh * HEAD_DIM + hd) * SEQ + s;      // [b,h,hd,s]
  const bf16* r = qkv + bs * (3 * N_EMBD);
  q[dst]   = f2bf(bf2f(r[d]) * 0.125f);            // fold 1/sqrt(64) into q
  k[dst]   = r[N_EMBD + d];
  vT[dstT] = r[2 * N_EMBD + d];
}

__global__ __launch_bounds__(256)
void merge_heads_k(const float* __restrict__ ah, bf16* __restrict__ out) {
  long i = (long)blockIdx.x * 256 + threadIdx.x;
  if (i >= (long)ROWS * N_EMBD) return;
  int  d  = (int)(i % N_EMBD);
  long bs = i / N_EMBD;
  int  s  = (int)(bs % SEQ);
  int  b  = (int)(bs / SEQ);
  int  h  = d / HEAD_DIM, hd = d % HEAD_DIM;
  out[i] = f2bf(ah[(((long)(b * N_HEAD + h)) * SEQ + s) * HEAD_DIM + hd]);
}

__global__ __launch_bounds__(256)
void softmax_causal_k(const float* __restrict__ S, bf16* __restrict__ P) {
  __shared__ float red[256];
  long row  = blockIdx.x;               // over B*H*S rows
  int  qpos = (int)(row % SEQ);
  int  len  = qpos + 1;
  const float* sr = S + row * SEQ;
  bf16*        pr = P + row * SEQ;
  int t = threadIdx.x;
  float mx = -1e30f;
  for (int j = t; j < len; j += 256) mx = fmaxf(mx, sr[j]);
  red[t] = mx; __syncthreads();
  for (int o = 128; o > 0; o >>= 1) { if (t < o) red[t] = fmaxf(red[t], red[t + o]); __syncthreads(); }
  mx = red[0]; __syncthreads();
  float sum = 0.f;
  for (int j = t; j < len; j += 256) sum += __expf(sr[j] - mx);
  red[t] = sum; __syncthreads();
  for (int o = 128; o > 0; o >>= 1) { if (t < o) red[t] += red[t + o]; __syncthreads(); }
  float inv = 1.0f / red[0];
  for (int j = t; j < SEQ; j += 256)
    pr[j] = f2bf(j < len ? __expf(sr[j] - mx) * inv : 0.0f);
}

// ---------- host orchestration ----------
static inline long cdiv(long a, long b) { return (a + b - 1) / b; }

extern "C" void kernel_launch(void* const* d_in, const int* in_sizes, int n_in,
                              void* d_out, int out_size, void* d_ws, size_t ws_size,
                              hipStream_t stream) {
  const float* x     = (const float*)d_in[0];
  const float* Wqkv  = (const float*)d_in[1];
  const float* bqkv  = (const float*)d_in[2];
  const float* Wo    = (const float*)d_in[3];
  const float* bo    = (const float*)d_in[4];
  const float* W1    = (const float*)d_in[5];
  const float* b1    = (const float*)d_in[6];
  const float* W2    = (const float*)d_in[7];
  const float* b2    = (const float*)d_in[8];
  const float* ln1g  = (const float*)d_in[9];
  const float* ln1b  = (const float*)d_in[10];
  const float* ln2g  = (const float*)d_in[11];
  const float* ln2b  = (const float*)d_in[12];
  const float* lnfg  = (const float*)d_in[13];
  const float* lnfb  = (const float*)d_in[14];
  const float* Wlm   = (const float*)d_in[15];
  float* out = (float*)d_out;

  // workspace carve-up (256B aligned)
  char* p = (char*)d_ws;
  auto alloc = [&](size_t bytes) -> void* {
    void* r = (void*)p;
    p += (bytes + 255) & ~(size_t)255;
    return r;
  };
  const size_t nWqkv = (size_t)N_LAYERS * N_EMBD * 3 * N_EMBD;
  const size_t nWo   = (size_t)N_LAYERS * N_EMBD * N_EMBD;
  const size_t nW1   = (size_t)N_LAYERS * N_EMBD * D_FF;
  const size_t nW2   = (size_t)N_LAYERS * D_FF * N_EMBD;
  const size_t nWlm  = (size_t)VOCAB * N_EMBD;

  // All weights stored TRANSPOSED bf16 ([N,K]) so every GEMM is NT with contiguous loads.
  bf16* wqkvT = (bf16*)alloc(nWqkv * 2);
  bf16* woT   = (bf16*)alloc(nWo * 2);
  bf16* w1T   = (bf16*)alloc(nW1 * 2);
  bf16* w2T   = (bf16*)alloc(nW2 * 2);
  bf16* wlm_b = (bf16*)alloc(nWlm * 2);   // Wlm is already [VOCAB, N_EMBD] == B^T form
  float* h    = (float*)alloc((size_t)ROWS * N_EMBD * 4);
  bf16* a_b   = (bf16*)alloc((size_t)ROWS * N_EMBD * 2);
  bf16* qkv_b = (bf16*)alloc((size_t)ROWS * 3 * N_EMBD * 2);
  bf16* q_b   = (bf16*)alloc((size_t)ROWS * N_EMBD * 2);
  bf16* k_b   = (bf16*)alloc((size_t)ROWS * N_EMBD * 2);
  bf16* vT_b  = (bf16*)alloc((size_t)ROWS * N_EMBD * 2);
  float* sc   = (float*)alloc((size_t)BATCH * N_HEAD * SEQ * SEQ * 4);
  bf16* pr_b  = (bf16*)alloc((size_t)BATCH * N_HEAD * SEQ * SEQ * 2);
  float* ahd  = (float*)alloc((size_t)ROWS * N_EMBD * 4);
  bf16* attn_b= (bf16*)alloc((size_t)ROWS * N_EMBD * 2);
  bf16* m_b   = (bf16*)alloc((size_t)ROWS * D_FF * 2);

  // weight prep: transpose+convert per layer (grid.z = layer)
  transpose_cvt_k<<<dim3(3 * N_EMBD / 32, N_EMBD / 32, N_LAYERS), 256, 0, stream>>>(
      Wqkv, wqkvT, N_EMBD, 3 * N_EMBD);
  transpose_cvt_k<<<dim3(N_EMBD / 32, N_EMBD / 32, N_LAYERS), 256, 0, stream>>>(
      Wo, woT, N_EMBD, N_EMBD);
  transpose_cvt_k<<<dim3(D_FF / 32, N_EMBD / 32, N_LAYERS), 256, 0, stream>>>(
      W1, w1T, N_EMBD, D_FF);
  transpose_cvt_k<<<dim3(N_EMBD / 32, D_FF / 32, N_LAYERS), 256, 0, stream>>>(
      W2, w2T, D_FF, N_EMBD);
  cvt_f32_bf16<<<cdiv(nWlm, 256), 256, 0, stream>>>(Wlm, wlm_b, (long)nWlm);

  // residual stream h = x
  const long nTok = (long)ROWS * N_EMBD;
  copy_f32<<<cdiv(nTok, 256), 256, 0, stream>>>(x, h, nTok);

  const long sQKh = (long)SEQ * HEAD_DIM;     // per (b,h) q/k/vT stride
  const long sSS  = (long)SEQ * SEQ;          // per (b,h) scores stride

  for (int l = 0; l < N_LAYERS; ++l) {
    const bf16* wq = wqkvT + (size_t)l * N_EMBD * 3 * N_EMBD;
    const bf16* wo = woT   + (size_t)l * N_EMBD * N_EMBD;
    const bf16* w1 = w1T   + (size_t)l * N_EMBD * D_FF;
    const bf16* w2 = w2T   + (size_t)l * D_FF * N_EMBD;

    // ln1 -> a
    layernorm_k<<<ROWS, 256, 0, stream>>>(h, ln1g + l * N_EMBD, ln1b + l * N_EMBD, a_b);

    // qkv = a @ WqkvT^T + bqkv  (bf16 out)
    gemm_nt<<<dim3(3 * N_EMBD / 64, ROWS / 32, 1), 32, 0, stream>>>(
        a_b, 0, N_EMBD, wq, 0, N_EMBD, bqkv + (size_t)l * 3 * N_EMBD,
        nullptr, 0, qkv_b, 0, 3 * N_EMBD, ROWS, 3 * N_EMBD, N_EMBD, 0, 0, 0);

    // split heads (scale q; v stored transposed)
    split_qkv_k<<<cdiv(nTok, 256), 256, 0, stream>>>(qkv_b, q_b, k_b, vT_b);

    // scores = q @ k^T   (batched over B*H; skip fully-masked tiles)
    gemm_nt<<<dim3(SEQ / 64, SEQ / 32, BATCH * N_HEAD), 32, 0, stream>>>(
        q_b, sQKh, HEAD_DIM, k_b, sQKh, HEAD_DIM, nullptr,
        sc, sSS, nullptr, 0, SEQ, SEQ, SEQ, HEAD_DIM, 0, 0, 1);

    // causal softmax -> bf16 probs
    softmax_causal_k<<<BATCH * N_HEAD * SEQ, 256, 0, stream>>>(sc, pr_b);

    // o = probs @ vT^T   (batched; K clamped to the causal frontier)
    gemm_nt<<<dim3(HEAD_DIM / 64, SEQ / 32, BATCH * N_HEAD), 32, 0, stream>>>(
        pr_b, sSS, SEQ, vT_b, sQKh, SEQ, nullptr,
        ahd, sQKh, nullptr, 0, HEAD_DIM, SEQ, HEAD_DIM, SEQ, 0, 0, 2);

    // merge heads -> bf16 [ROWS, D]
    merge_heads_k<<<cdiv(nTok, 256), 256, 0, stream>>>(ahd, attn_b);

    // h += attn @ WoT^T + bo
    gemm_nt<<<dim3(N_EMBD / 64, ROWS / 32, 1), 32, 0, stream>>>(
        attn_b, 0, N_EMBD, wo, 0, N_EMBD, bo + (size_t)l * N_EMBD,
        h, 0, nullptr, 0, N_EMBD, ROWS, N_EMBD, N_EMBD, 1, 0, 0);

    // ln2 -> a
    layernorm_k<<<ROWS, 256, 0, stream>>>(h, ln2g + l * N_EMBD, ln2b + l * N_EMBD, a_b);

    // m = gelu(a @ W1T^T + b1)  (bf16 out)
    gemm_nt<<<dim3(D_FF / 64, ROWS / 32, 1), 32, 0, stream>>>(
        a_b, 0, N_EMBD, w1, 0, N_EMBD, b1 + (size_t)l * D_FF,
        nullptr, 0, m_b, 0, D_FF, ROWS, D_FF, N_EMBD, 0, 1, 0);

    // h += m @ W2T^T + b2
    gemm_nt<<<dim3(N_EMBD / 64, ROWS / 32, 1), 32, 0, stream>>>(
        m_b, 0, D_FF, w2, 0, D_FF, b2 + (size_t)l * N_EMBD,
        h, 0, nullptr, 0, N_EMBD, ROWS, N_EMBD, D_FF, 1, 0, 0);
  }

  // final LN + LM head: out = ln(h) @ Wlm^T
  layernorm_k<<<ROWS, 256, 0, stream>>>(h, lnfg, lnfb, a_b);
  gemm_nt<<<dim3((VOCAB + 63) / 64, ROWS / 32, 1), 32, 0, stream>>>(
      a_b, 0, N_EMBD, wlm_b, 0, N_EMBD, nullptr,
      out, 0, nullptr, 0, VOCAB, ROWS, VOCAB, N_EMBD, 0, 0, 0);
}
